// HNUNet_44976897523754
// MI455X (gfx1250) — matrix operand order repaired and checked by reading, not compiled, not run on hardware
//
#include <hip/hip_runtime.h>
#include <cmath>

// ---------------------------------------------------------------------------
// Hyperbolic U-Net (Poincare ball) forward for MI455X / gfx1250.
// wave32 everywhere; matmuls via v_wmma_f32_16x16x32_bf16 with vectorized
// LDS staging (b128) and GLOBAL_LOAD_ASYNC_TO_LDS_B128 when available.
// ---------------------------------------------------------------------------

#define EPSF  1e-12f
#define MAXNF (1.0f - 1e-5f)

typedef __attribute__((ext_vector_type(16))) __bf16 v16bf;
typedef __attribute__((ext_vector_type(8)))  __bf16 v8bf;
typedef __attribute__((ext_vector_type(8)))  float  v8f;
typedef __attribute__((ext_vector_type(4)))  int    v4i;

#if defined(__has_builtin)
#if __has_builtin(__builtin_amdgcn_global_load_async_to_lds_b128)
#define HAVE_ASYNC_LDS 1
#endif
#if __has_builtin(__builtin_amdgcn_s_wait_asynccnt)
#define HAVE_WAIT_ASYNC_BUILTIN 1
#endif
#endif

#define GAS __attribute__((address_space(1)))
#define LAS __attribute__((address_space(3)))

__device__ __forceinline__ void wait_async0() {
#ifdef HAVE_ASYNC_LDS
#ifdef HAVE_WAIT_ASYNC_BUILTIN
  __builtin_amdgcn_s_wait_asynccnt(0);
#else
  asm volatile("s_wait_asynccnt 0" ::: "memory");
#endif
#endif
}

// ---------------- wave helpers (wave32) ----------------
__device__ __forceinline__ float wsum(float v) {
#pragma unroll
  for (int o = 16; o > 0; o >>= 1) v += __shfl_xor(v, o, 32);
  return v;
}

template <int NC>
__device__ __forceinline__ float sqn(const float (&v)[NC]) {
  float s = 0.f;
#pragma unroll
  for (int i = 0; i < NC; ++i) s += v[i] * v[i];
  return wsum(s);
}

template <int NC>
__device__ __forceinline__ void ldv(const float* p, int C, float (&v)[NC]) {
  int lane = threadIdx.x & 31;
#pragma unroll
  for (int i = 0; i < NC; ++i) { int ci = lane + i * 32; v[i] = (ci < C) ? p[ci] : 0.f; }
}

template <int NC>
__device__ __forceinline__ void stv(float* p, int C, const float (&v)[NC]) {
  int lane = threadIdx.x & 31;
#pragma unroll
  for (int i = 0; i < NC; ++i) { int ci = lane + i * 32; if (ci < C) p[ci] = v[i]; }
}

// factor for logmap0: atanh(min(sc*n,MAXN))/(sc*n)
__device__ __forceinline__ float log0f_(float sq, float sc) {
  float n = sqrtf(fmaxf(sq, EPSF));
  return atanhf(fminf(sc * n, MAXNF)) / (sc * n);
}
// factor for expmap0 with projection clamp
__device__ __forceinline__ float exp0f_(float sq, float sc) {
  float n = sqrtf(fmaxf(sq, EPSF));
  float th = fminf(tanhf(sc * n), MAXNF);
  return th / (sc * n);
}

// mobius_add(sa*a, sb*b) -> o  (o may alias b)
template <int NC>
__device__ __forceinline__ void mob(const float (&a)[NC], float sa,
                                    const float (&b)[NC], float sb,
                                    float c, float (&o)[NC]) {
  float ab = 0.f, a2 = 0.f, b2 = 0.f;
#pragma unroll
  for (int i = 0; i < NC; ++i) {
    float ai = sa * a[i], bi = sb * b[i];
    ab += ai * bi; a2 += ai * ai; b2 += bi * bi;
  }
  ab = wsum(ab); a2 = wsum(a2); b2 = wsum(b2);
  float k1 = 1.f + 2.f * c * ab + c * b2;
  float k2 = 1.f - c * a2;
  float den = fmaxf(1.f + 2.f * c * ab + c * c * a2 * b2, EPSF);
#pragma unroll
  for (int i = 0; i < NC; ++i) o[i] = (k1 * sa * a[i] + k2 * sb * b[i]) / den;
}

// ---------------- small utility kernels ----------------
__global__ void k_zero(float* p, int n) {
  for (int i = blockIdx.x * blockDim.x + threadIdx.x; i < n; i += blockDim.x * gridDim.x)
    p[i] = 0.f;
}

__global__ void k_colnorm(const float* __restrict__ z, float* __restrict__ zn, int K, int N) {
  int n = blockIdx.x * blockDim.x + threadIdx.x;
  if (n >= N) return;
  float s = 0.f;
  for (int k = 0; k < K; ++k) { float v = z[(size_t)k * N + n]; s += v * v; }
  zn[n] = fmaxf(sqrtf(s), EPSF);
}

// z[K,N] f32 -> zT[N,Kpad] bf16 (transposed, zero-padded to Kpad)
__global__ void k_cvt_zT(const float* __restrict__ z, __bf16* __restrict__ zT,
                         int K, int N, int Kpad) {
  long long tot = (long long)N * Kpad;
  for (long long i = blockIdx.x * (long long)blockDim.x + threadIdx.x; i < tot;
       i += (long long)blockDim.x * gridDim.x) {
    int n = (int)(i / Kpad);
    int k = (int)(i - (long long)n * Kpad);
    zT[i] = (k < K) ? (__bf16)z[(size_t)k * N + n] : (__bf16)0.f;
  }
}

// input NCHW f32 -> NHWC ball via expmap0 (C <= 32)
__global__ void k_in_expmap0(const float* __restrict__ x, float* __restrict__ out,
                             int B, int C, int H, int W, const float* cptr) {
  int P = B * H * W;
  int pix = blockIdx.x * 8 + (threadIdx.x >> 5);
  if (pix >= P) return;
  int b = pix / (H * W), rem = pix - b * H * W, h = rem / W, w = rem - h * W;
  int lane = threadIdx.x & 31;
  float c = *cptr, sc = sqrtf(c);
  float v = (lane < C) ? x[(((size_t)b * C + lane) * H + h) * W + w] : 0.f;
  float f = exp0f_(wsum(v * v), sc);
  if (lane < C) out[(size_t)pix * C + lane] = v * f;
}

// ball NHWC -> tangent NHWC
template <int NC>
__global__ void k_logmap0(const float* __restrict__ in, float* __restrict__ out,
                          int P, int C, const float* cptr) {
  int pix = blockIdx.x * 8 + (threadIdx.x >> 5);
  if (pix >= P) return;
  float c = *cptr, sc = sqrtf(c);
  float v[NC];
  ldv<NC>(in + (size_t)pix * C, C, v);
  float f = log0f_(sqn<NC>(v), sc);
#pragma unroll
  for (int i = 0; i < NC; ++i) v[i] *= f;
  stv<NC>(out + (size_t)pix * C, C, v);
}

// hyperbolic 2x2 maxpool (logmap0 -> elementwise max -> expmap0)
template <int NC>
__global__ void k_maxpool(const float* __restrict__ in, float* __restrict__ out,
                          int B, int Ho, int Wo, int C, const float* cptr) {
  int P = B * Ho * Wo;
  int pix = blockIdx.x * 8 + (threadIdx.x >> 5);
  if (pix >= P) return;
  int b = pix / (Ho * Wo), rem = pix - b * Ho * Wo, h = rem / Wo, w = rem - h * Wo;
  int Hi = 2 * Ho, Wi = 2 * Wo;
  float c = *cptr, sc = sqrtf(c);
  int lane = threadIdx.x & 31;
  float t[NC];
#pragma unroll
  for (int i = 0; i < NC; ++i) t[i] = -3.4e38f;
  for (int dy = 0; dy < 2; ++dy)
    for (int dx = 0; dx < 2; ++dx) {
      const float* p = in + (((size_t)b * Hi + (2 * h + dy)) * Wi + (2 * w + dx)) * C;
      float v[NC];
      ldv<NC>(p, C, v);
      float f = log0f_(sqn<NC>(v), sc);
#pragma unroll
      for (int i = 0; i < NC; ++i) t[i] = fmaxf(t[i], v[i] * f);
    }
#pragma unroll
  for (int i = 0; i < NC; ++i) { int ci = lane + i * 32; if (ci >= C) t[i] = 0.f; }
  float ef = exp0f_(sqn<NC>(t), sc);
#pragma unroll
  for (int i = 0; i < NC; ++i) t[i] *= ef;
  stv<NC>(out + (size_t)pix * C, C, t);
}

// hyperbolic bilinear 2x upsample (half-pixel, edge clamp)
template <int NC>
__global__ void k_upsample(const float* __restrict__ in, float* __restrict__ out,
                           int B, int Hi, int Wi, int C, const float* cptr) {
  int Ho = 2 * Hi, Wo = 2 * Wi;
  int P = B * Ho * Wo;
  int pix = blockIdx.x * 8 + (threadIdx.x >> 5);
  if (pix >= P) return;
  int b = pix / (Ho * Wo), rem = pix - b * Ho * Wo, h = rem / Wo, w = rem - h * Wo;
  float c = *cptr, sc = sqrtf(c);
  float sy = 0.5f * (h + 0.5f) - 0.5f, sx = 0.5f * (w + 0.5f) - 0.5f;
  int h0 = (int)floorf(sy), w0 = (int)floorf(sx);
  float fy = sy - (float)h0, fx = sx - (float)w0;
  int h1 = min(max(h0 + 1, 0), Hi - 1), w1 = min(max(w0 + 1, 0), Wi - 1);
  h0 = min(max(h0, 0), Hi - 1); w0 = min(max(w0, 0), Wi - 1);
  float wgt[4] = {(1.f - fy) * (1.f - fx), (1.f - fy) * fx, fy * (1.f - fx), fy * fx};
  int hh[4] = {h0, h0, h1, h1}, wwv[4] = {w0, w1, w0, w1};
  float t[NC];
#pragma unroll
  for (int i = 0; i < NC; ++i) t[i] = 0.f;
  for (int k = 0; k < 4; ++k) {
    const float* p = in + (((size_t)b * Hi + hh[k]) * Wi + wwv[k]) * C;
    float v[NC];
    ldv<NC>(p, C, v);
    float f = log0f_(sqn<NC>(v), sc);
#pragma unroll
    for (int i = 0; i < NC; ++i) t[i] += wgt[k] * f * v[i];
  }
  float ef = exp0f_(sqn<NC>(t), sc);
#pragma unroll
  for (int i = 0; i < NC; ++i) t[i] *= ef;
  stv<NC>(out + (size_t)pix * C, C, t);
}

// hyperbolic beta-concat: [s1*logmap0(a) , s2*logmap0(b)] -> expmap0
template <int NC>
__global__ void k_cat(const float* __restrict__ a, const float* __restrict__ bsrc,
                      float* __restrict__ out, int P, int C1, int C2,
                      float s1, float s2, const float* cptr) {
  int pix = blockIdx.x * 8 + (threadIdx.x >> 5);
  if (pix >= P) return;
  int lane = threadIdx.x & 31;
  int Ct = C1 + C2;
  float c = *cptr, sc = sqrtf(c);
  float v[NC];
  float q1 = 0.f, q2 = 0.f;
#pragma unroll
  for (int i = 0; i < NC; ++i) {
    int ci = lane + i * 32;
    float val = 0.f;
    if (ci < C1)      { val = a[(size_t)pix * C1 + ci];            q1 += val * val; }
    else if (ci < Ct) { val = bsrc[(size_t)pix * C2 + (ci - C1)];  q2 += val * val; }
    v[i] = val;
  }
  q1 = wsum(q1); q2 = wsum(q2);
  float f1 = log0f_(q1, sc) * s1, f2 = log0f_(q2, sc) * s2;
#pragma unroll
  for (int i = 0; i < NC; ++i) { int ci = lane + i * 32; v[i] *= (ci < C1) ? f1 : f2; }
  float ef = exp0f_(sqn<NC>(v), sc);
#pragma unroll
  for (int i = 0; i < NC; ++i) v[i] *= ef;
  stv<NC>(out + (size_t)pix * Ct, Ct, v);
}

// ---------------- hyperbolic batch-norm ----------------
// stats layout: [0..C-1]=sum(lam*x), [C]=sum(lam-1), [C+1]=sum(pdist^2)
template <int NC>
__global__ void k_bn_reduce(const float* __restrict__ xin, float* __restrict__ stats,
                            int P, int C, const float* cptr) {
  int pix = blockIdx.x * 8 + (threadIdx.x >> 5);
  if (pix >= P) return;
  int lane = threadIdx.x & 31;
  float c = *cptr;
  float x[NC];
  ldv<NC>(xin + (size_t)pix * C, C, x);
  float lam = 2.f / fmaxf(1.f - c * sqn<NC>(x), EPSF);
#pragma unroll
  for (int i = 0; i < NC; ++i) { int ci = lane + i * 32; if (ci < C) atomicAdd(&stats[ci], lam * x[i]); }
  if (lane == 0) atomicAdd(&stats[C], lam - 1.f);
}

template <int NC>
__global__ void k_bn_mean(const float* __restrict__ stats, float* __restrict__ meanb,
                          int C, const float* cptr) {
  float c = *cptr, sc = sqrtf(c);
  float m[NC];
  ldv<NC>(stats, C, m);
  float den = fmaxf(stats[C], EPSF);
#pragma unroll
  for (int i = 0; i < NC; ++i) m[i] /= den;
  float n = sqrtf(fmaxf(sqn<NC>(m), EPSF));
  float f = tanhf(0.5f * atanhf(fminf(sc * n, MAXNF))) / (sc * n);
  float mx = MAXNF / sc, nn = f * n;
  if (nn > mx) f *= mx / nn;
#pragma unroll
  for (int i = 0; i < NC; ++i) m[i] *= f;
  stv<NC>(meanb, C, m);
}

template <int NC>
__global__ void k_bn_var(const float* __restrict__ xin, const float* __restrict__ meanb,
                         float* __restrict__ stats, int P, int C, const float* cptr) {
  int pix = blockIdx.x * 8 + (threadIdx.x >> 5);
  if (pix >= P) return;
  float c = *cptr, sc = sqrtf(c);
  float x[NC], mn[NC], u[NC];
  ldv<NC>(xin + (size_t)pix * C, C, x);
  ldv<NC>(meanb, C, mn);
  mob<NC>(x, -1.f, mn, 1.f, c, u);               // (-x) (+) mean
  float n = sqrtf(fmaxf(sqn<NC>(u), EPSF));
  float d = (2.f / sc) * atanhf(fminf(sc * n, MAXNF));
  if ((threadIdx.x & 31) == 0) atomicAdd(&stats[C + 1], d * d);
}

// apply: v=logmap(mean,x); transp(mean,b,v); scale; expmap(b, .); fused h_relu
template <int NC>
__global__ void k_bn_apply(const float* __restrict__ xin, const float* __restrict__ meanb,
                           const float* __restrict__ bvec, const float* __restrict__ wptr,
                           const float* __restrict__ stats, float* __restrict__ outb,
                           int P, int C, const float* cptr) {
  int pix = blockIdx.x * 8 + (threadIdx.x >> 5);
  if (pix >= P) return;
  float c = *cptr, sc = sqrtf(c);
  float x[NC], mn[NC], bp[NC], t1[NC];
  ldv<NC>(xin + (size_t)pix * C, C, x);
  ldv<NC>(meanb, C, mn);
  ldv<NC>(bvec, C, bp);
  float lam_m = 2.f / fmaxf(1.f - c * sqn<NC>(mn), EPSF);
  float lam_b = 2.f / fmaxf(1.f - c * sqn<NC>(bp), EPSF);
  // v = logmap(mean, x)
  mob<NC>(mn, -1.f, x, 1.f, c, t1);
  float un = sqrtf(fmaxf(sqn<NC>(t1), EPSF));
  float vf = (2.f / (sc * lam_m)) * atanhf(fminf(sc * un, MAXNF)) / un;
#pragma unroll
  for (int i = 0; i < NC; ++i) t1[i] *= vf;
  // gyration(b, -mean, v)
  mob<NC>(mn, -1.f, t1, 1.f, c, x);   // (-mean) (+) v
  mob<NC>(bp, 1.f, x, 1.f, c, x);     // b (+) ...
  mob<NC>(bp, 1.f, mn, -1.f, c, t1);  // s = b (+) (-mean)
  mob<NC>(t1, -1.f, x, 1.f, c, x);    // (-s) (+) ...
  float var = stats[C + 1] / (float)P;
  float g = (lam_m / lam_b) * sqrtf((*wptr) / (var + 1e-6f));
#pragma unroll
  for (int i = 0; i < NC; ++i) x[i] *= g;
  // expmap(b, vs)
  float vn = sqrtf(fmaxf(sqn<NC>(x), EPSF));
  float sf = tanhf(sc * lam_b * vn * 0.5f) / (sc * vn);
#pragma unroll
  for (int i = 0; i < NC; ++i) x[i] *= sf;
  mob<NC>(bp, 1.f, x, 1.f, c, x);
  float rn = sqrtf(fmaxf(sqn<NC>(x), EPSF));
  float mx = MAXNF / sc;
  float cf = (rn > mx) ? mx / rn : 1.f;
#pragma unroll
  for (int i = 0; i < NC; ++i) x[i] *= cf;
  // fused h_relu: expmap0(relu(logmap0(x)))
  float lf = log0f_(sqn<NC>(x), sc);
#pragma unroll
  for (int i = 0; i < NC; ++i) x[i] = fmaxf(x[i] * lf, 0.f);
  float ef = exp0f_(sqn<NC>(x), sc);
#pragma unroll
  for (int i = 0; i < NC; ++i) x[i] *= ef;
  stv<NC>(outb + (size_t)pix * C, C, x);
}

// ---------------- fused im2col + bf16 WMMA GEMM + poincare_fc ----------------
// A[M=P,K] gathered from tangent NHWC; B = zT bf16 [N,Kpad] (pre-transposed).
// NCH = compile-time number of 128-col chunks (straight-line accumulators).
template <int NCH>
__global__ void __launch_bounds__(256)
k_fc_gemm(const float* __restrict__ t, const __bf16* __restrict__ zT,
          const float* __restrict__ zn, const float* __restrict__ r,
          const float* __restrict__ cptr, float scale,
          int B, int H, int W, int Cin, int N, int ksz, int pad, int Kpad,
          float* __restrict__ out) {
  __shared__ __bf16 Ash[16][32];    // row-major M x K-chunk
  __shared__ __bf16 Bsh[128][32];   // fragment-major: col x K-chunk
  __shared__ float rowsq[16];
  __shared__ float wsqs[16];

  const int tid = threadIdx.x;
  const int wave = tid >> 5, lane = tid & 31;
  const int P = B * H * W;
  const int m0 = blockIdx.x * 16;
  const int K = Cin * ksz * ksz;
  const int Ksteps = Kpad >> 5;

  if (tid < 16) { rowsq[tid] = 0.f; wsqs[tid] = 0.f; }
  __syncthreads();

  const v8f vzero = {0.f, 0.f, 0.f, 0.f, 0.f, 0.f, 0.f, 0.f};
  v8f acc[NCH];
#pragma unroll
  for (int q = 0; q < NCH; ++q) acc[q] = vzero;

  // B staging geometry: thread -> (col, 16-element k-half)
  const int bcol = tid >> 1;
  const int bkb  = (tid & 1) << 4;

  for (int kt = 0; kt < Ksteps; ++kt) {
    // ---- stage A tile 16x32 (gathered im2col, f32 -> bf16) ----
#pragma unroll
    for (int e2 = 0; e2 < 2; ++e2) {
      int e = tid * 2 + e2;                 // 0..511
      int mrow = e >> 5;
      int kk = (kt << 5) + (e & 31);
      float val = 0.f;
      int pix = m0 + mrow;
      if (pix < P && kk < K) {
        int b = pix / (H * W), rem = pix - b * H * W, h = rem / W, w = rem - h * W;
        int ci = kk / (ksz * ksz);
        int rr = kk - ci * ksz * ksz;
        int ry = rr / ksz;
        int dy = ry - pad, dx = rr - ry * ksz - pad;
        int h2 = h + dy, w2 = w + dx;
        if (h2 >= 0 && h2 < H && w2 >= 0 && w2 < W)
          val = t[(((size_t)b * H + h2) * W + w2) * Cin + ci];
      }
      Ash[mrow][e & 31] = (__bf16)val;
      if (val != 0.f) atomicAdd(&rowsq[mrow], val * val);
    }
    __syncthreads();
    // A fragment (same for all waves): 2x ds_load_b128 + concat
    v16bf afrag;
    {
      int mrow = lane & 15;
      int kb = (lane < 16) ? 0 : 8;
      v8bf alo = *(const v8bf*)&Ash[mrow][kb];
      v8bf ahi = *(const v8bf*)&Ash[mrow][kb + 16];
      afrag = __builtin_shufflevector(alo, ahi, 0, 1, 2, 3, 4, 5, 6, 7,
                                      8, 9, 10, 11, 12, 13, 14, 15);
    }
#pragma unroll
    for (int q = 0; q < NCH; ++q) {
      // ---- stage B tile: 128 cols x 32 k, 32 contiguous bytes per thread ----
      int col = (q << 7) + bcol;
      __bf16* dst = &Bsh[bcol][bkb];
      if (col < N) {
        const __bf16* src = zT + (size_t)col * Kpad + (kt << 5) + bkb;
#ifdef HAVE_ASYNC_LDS
        __builtin_amdgcn_global_load_async_to_lds_b128(
            (GAS v4i*)src, (LAS v4i*)dst, 0, 0);
        __builtin_amdgcn_global_load_async_to_lds_b128(
            (GAS v4i*)(src + 8), (LAS v4i*)(dst + 8), 0, 0);
#else
        *(v8bf*)dst       = *(const v8bf*)src;
        *(v8bf*)(dst + 8) = *(const v8bf*)(src + 8);
#endif
      } else {
        v8bf z8;
#pragma unroll
        for (int j = 0; j < 8; ++j) z8[j] = (__bf16)0.f;
        *(v8bf*)dst       = z8;
        *(v8bf*)(dst + 8) = z8;
      }
      wait_async0();
      __syncthreads();
      // B fragment: col = wave*16 + lane%16; K 8-blocks {kb, kb+16} by lane group
      v16bf bfrag;
      {
        int bc = (wave << 4) + (lane & 15);
        int kb = (lane < 16) ? 0 : 8;
        v8bf blo = *(const v8bf*)&Bsh[bc][kb];
        v8bf bhi = *(const v8bf*)&Bsh[bc][kb + 16];
        bfrag = __builtin_shufflevector(blo, bhi, 0, 1, 2, 3, 4, 5, 6, 7,
                                        8, 9, 10, 11, 12, 13, 14, 15);
      }
      acc[q] = __builtin_amdgcn_wmma_f32_16x16x32_bf16(
          false, afrag, false, bfrag, (short)0, acc[q], false, false);
      __syncthreads();
    }
    if (kt + 1 < Ksteps && lane == 0)
      __builtin_prefetch(&zT[(size_t)(kt + 1) << 5], 0, 1);
  }
  __syncthreads();

  const float c = *cptr;
  const float sc = sqrtf(c);
  const int coltile = (wave << 4) + (lane & 15);
  const int rbase = (lane >= 16) ? 8 : 0;

  // per-row factors (depend only on rowsq)
  float rowf[8], rowlam[8];
#pragma unroll
  for (int j = 0; j < 8; ++j) {
    int trow = j + rbase;
    float npt = sqrtf(fmaxf(rowsq[trow], EPSF));
    float n = scale * npt;                        // ||pt*scale||
    float th = fminf(tanhf(sc * n), MAXNF);       // projected expmap0 tanh
    rowf[j] = th / (sc * n) * scale;              // x.z = rowf * (pt.z)
    float yn = th / sc;
    rowlam[j] = 2.f / fmaxf(1.f - c * yn * yn, EPSF);
  }

  // pass 1: w values + per-row sum(w^2)
  float wv[NCH][8];
#pragma unroll
  for (int q = 0; q < NCH; ++q) {
    int col = (q << 7) + coltile;
    float znc = (col < N) ? fmaxf(zn[col], EPSF) : 1.f;
    float rc = (col < N) ? r[col] : 0.f;
    float tcr = 2.f * sc * rc;
    float chh = coshf(tcr), shh = sinhf(tcr);
#pragma unroll
    for (int j = 0; j < 8; ++j) {
      int trow = j + rbase;
      float lamx = rowlam[j];
      float xzn = rowf[j] * acc[q][j] / znc;
      float arg = sc * lamx * xzn * chh - (lamx - 1.f) * shh;
      float v = (2.f * znc / sc) * asinhf(arg);
      float wvv = sinhf(sc * v) / sc;
      if (col >= N || (m0 + trow) >= P) wvv = 0.f;
      wv[q][j] = wvv;
      if (wvv != 0.f) atomicAdd(&wsqs[trow], wvv * wvv);
    }
  }
  __syncthreads();

  // pass 2: w/(1+sqrt(1+c|w|^2)) + projection (all per-row scalars)
#pragma unroll
  for (int q = 0; q < NCH; ++q) {
    int col = (q << 7) + coltile;
#pragma unroll
    for (int j = 0; j < 8; ++j) {
      int trow = j + rbase;
      int pix = m0 + trow;
      if (col < N && pix < P) {
        float ws = wsqs[trow];
        float den = 1.f + sqrtf(1.f + c * ws);
        float g = 1.f / den;
        float dn = sqrtf(fmaxf(ws, EPSF)) * g;
        float mx = MAXNF / sc;
        float cf = (dn > mx) ? mx / dn : 1.f;
        out[(size_t)pix * N + col] = wv[q][j] * g * cf;
      }
    }
  }
}

// final: ball NHWC [P,4] -> logmap0 -> NCHW f32 output
__global__ void k_final_out(const float* __restrict__ in, float* __restrict__ out,
                            int B, int H, int W, int C, const float* cptr) {
  int P = B * H * W;
  int pix = blockIdx.x * blockDim.x + threadIdx.x;
  if (pix >= P) return;
  int b = pix / (H * W), rem = pix - b * H * W, h = rem / W, w = rem - h * W;
  float c = *cptr, sc = sqrtf(c);
  float v[4];
  float sq = 0.f;
  for (int ci = 0; ci < 4; ++ci) { v[ci] = in[(size_t)pix * C + ci]; sq += v[ci] * v[ci]; }
  float n = sqrtf(fmaxf(sq, EPSF));
  float f = atanhf(fminf(sc * n, MAXNF)) / (sc * n);
  for (int ci = 0; ci < 4; ++ci)
    out[(((size_t)b * C + ci) * H + h) * W + w] = v[ci] * f;
}

// ---------------- host orchestration ----------------
static inline double beta_h(double n) {
  return exp(::lgamma(n / 2.0) + ::lgamma(0.5) - ::lgamma((n + 1.0) / 2.0));
}

#define DNC(NCV, ...)                                          \
  do {                                                         \
    int _n = (NCV);                                            \
    if (_n <= 1)       { constexpr int KNC = 1;  __VA_ARGS__; } \
    else if (_n <= 2)  { constexpr int KNC = 2;  __VA_ARGS__; } \
    else if (_n <= 4)  { constexpr int KNC = 4;  __VA_ARGS__; } \
    else if (_n <= 8)  { constexpr int KNC = 8;  __VA_ARGS__; } \
    else if (_n <= 16) { constexpr int KNC = 16; __VA_ARGS__; } \
    else               { constexpr int KNC = 24; __VA_ARGS__; } \
  } while (0)

struct Blk { const float *b1, *w1, *b2, *w2, *r1, *z1, *r2, *z2; };

extern "C" void kernel_launch(void* const* d_in, const int* in_sizes, int n_in,
                              void* d_out, int out_size, void* d_ws, size_t ws_size,
                              hipStream_t stream) {
  (void)in_sizes; (void)out_size; (void)ws_size;
  const int NB = 2;
  // JAX pytree flattening order (dicts sorted alphabetically):
  // d_in[0]=x; [1..72]=blocks {conv0_0,conv0_4,conv1_0,conv1_3,conv2_0,conv2_2,
  //   conv3_0,conv3_1,conv4_0} x {bn1.b,bn1.w,bn2.b,bn2.w,conv1.r,conv1.z,conv2.r,conv2.z};
  // [73]=c; [74]=final.r; [75]=final.z
  auto IN = [&](int i) -> const float* { return (const float*)d_in[(i < n_in) ? i : 0]; };
  const float* x_nchw = IN(0);
  const float* cptr = IN(73);
  const float* final_r = IN(74);
  const float* final_z = IN(75);
  auto getb = [&](int bi) {
    int base = 1 + bi * 8;
    Blk b;
    b.b1 = IN(base + 0); b.w1 = IN(base + 1);
    b.b2 = IN(base + 2); b.w2 = IN(base + 3);
    b.r1 = IN(base + 4); b.z1 = IN(base + 5);
    b.r2 = IN(base + 6); b.z2 = IN(base + 7);
    return b;
  };

  // arenas over d_ws: persistent region then transient region (reset per stage)
  char* wsb = (char*)d_ws;
  size_t pOff = 0;
  const size_t TBASE = 72ull << 20;
  size_t tOff = TBASE;
  auto pall = [&](size_t bytes) -> void* { void* p = wsb + pOff; pOff += (bytes + 255) & ~(size_t)255; return p; };
  auto tall = [&](size_t bytes) -> void* { void* p = wsb + tOff; tOff += (bytes + 255) & ~(size_t)255; return p; };
  auto resetT = [&]() { tOff = TBASE; };

  auto run_fc = [&](const float* tbuf, const float* z, const float* r,
                    int Cin, int N, int ksz, int pad, int H, int W,
                    float scale, float* outb) {
    int K = Cin * ksz * ksz;
    int Kpad = (K + 31) & ~31;
    int P = NB * H * W;
    int nch = (N + 127) >> 7;
    float* zn = (float*)tall((size_t)N * 4);
    __bf16* zT = (__bf16*)tall((size_t)N * Kpad * 2);
    k_colnorm<<<dim3((N + 63) / 64), dim3(64), 0, stream>>>(z, zn, K, N);
    k_cvt_zT<<<dim3(512), dim3(256), 0, stream>>>(z, zT, K, N, Kpad);
    dim3 g((P + 15) / 16), blk(256);
    if (nch <= 1)
      k_fc_gemm<1><<<g, blk, 0, stream>>>(tbuf, zT, zn, r, cptr, scale, NB, H, W, Cin, N, ksz, pad, Kpad, outb);
    else if (nch == 2)
      k_fc_gemm<2><<<g, blk, 0, stream>>>(tbuf, zT, zn, r, cptr, scale, NB, H, W, Cin, N, ksz, pad, Kpad, outb);
    else
      k_fc_gemm<4><<<g, blk, 0, stream>>>(tbuf, zT, zn, r, cptr, scale, NB, H, W, Cin, N, ksz, pad, Kpad, outb);
  };

  auto run_bn = [&](const float* xin, const float* bvec, const float* wptr,
                    int C, int P, float* outb) {
    int nc = (C + 31) / 32;
    float* stats = (float*)tall((size_t)(C + 2) * 4);
    float* meanb = (float*)tall((size_t)C * 4);
    k_zero<<<dim3(1), dim3(256), 0, stream>>>(stats, C + 2);
    DNC(nc, k_bn_reduce<KNC><<<dim3((P + 7) / 8), dim3(256), 0, stream>>>(xin, stats, P, C, cptr));
    DNC(nc, k_bn_mean<KNC><<<dim3(1), dim3(32), 0, stream>>>(stats, meanb, C, cptr));
    DNC(nc, k_bn_var<KNC><<<dim3((P + 7) / 8), dim3(256), 0, stream>>>(xin, meanb, stats, P, C, cptr));
    DNC(nc, k_bn_apply<KNC><<<dim3((P + 7) / 8), dim3(256), 0, stream>>>(
                xin, meanb, bvec, wptr, stats, outb, P, C, cptr));
  };

  auto run_vgg = [&](const float* ball_in, int Cin, int Cmid, int Cout,
                     int H, int W, const Blk& bp, float* ball_out) {
    int P = NB * H * W;
    float s1 = (float)(beta_h(9.0 * Cin) / beta_h((double)Cin));
    float s2 = (float)(beta_h(9.0 * Cmid) / beta_h((double)Cmid));
    float* t1 = (float*)tall((size_t)P * Cin * 4);
    DNC((Cin + 31) / 32,
        k_logmap0<KNC><<<dim3((P + 7) / 8), dim3(256), 0, stream>>>(ball_in, t1, P, Cin, cptr));
    float* mid = (float*)tall((size_t)P * Cmid * 4);
    run_fc(t1, bp.z1, bp.r1, Cin, Cmid, 3, 1, H, W, s1, mid);
    float* mid2 = (float*)tall((size_t)P * Cmid * 4);
    run_bn(mid, bp.b1, bp.w1, Cmid, P, mid2);
    float* t2 = (float*)tall((size_t)P * Cmid * 4);
    DNC((Cmid + 31) / 32,
        k_logmap0<KNC><<<dim3((P + 7) / 8), dim3(256), 0, stream>>>(mid2, t2, P, Cmid, cptr));
    float* out2 = (float*)tall((size_t)P * Cout * 4);
    run_fc(t2, bp.z2, bp.r2, Cmid, Cout, 3, 1, H, W, s2, out2);
    run_bn(out2, bp.b2, bp.w2, Cout, P, ball_out);
  };

  auto run_cat = [&](const float* a, const float* b, int P, int C1, int C2, float* outb) {
    double bN = beta_h((double)(C1 + C2));
    float s1 = (float)(bN / beta_h((double)C1));
    float s2 = (float)(bN / beta_h((double)C2));
    DNC((C1 + C2 + 31) / 32,
        k_cat<KNC><<<dim3((P + 7) / 8), dim3(256), 0, stream>>>(a, b, outb, P, C1, C2, s1, s2, cptr));
  };

  const int P192 = NB * 192 * 192, P96 = NB * 96 * 96, P48 = NB * 48 * 48,
            P24 = NB * 24 * 24, P12 = NB * 12 * 12;

  // persistent buffers
  float* ballx = (float*)pall((size_t)P192 * 3 * 4);
  float* x0 = (float*)pall((size_t)P192 * 32 * 4);
  float* x1 = (float*)pall((size_t)P96 * 64 * 4);
  float* x2 = (float*)pall((size_t)P48 * 128 * 4);
  float* x3 = (float*)pall((size_t)P24 * 256 * 4);
  float* x4 = (float*)pall((size_t)P12 * 512 * 4);
  float* x3n = (float*)pall((size_t)P24 * 256 * 4);
  float* x2n = (float*)pall((size_t)P48 * 128 * 4);
  float* x1n = (float*)pall((size_t)P96 * 64 * 4);
  float* x0n = (float*)pall((size_t)P192 * 32 * 4);

  // input: NCHW -> NHWC ball
  k_in_expmap0<<<dim3((P192 + 7) / 8), dim3(256), 0, stream>>>(x_nchw, ballx, NB, 3, 192, 192, cptr);

  // encoder
  resetT();
  run_vgg(ballx, 3, 32, 32, 192, 192, getb(0), x0);              // conv0_0
  resetT();
  { float* p = (float*)tall((size_t)P96 * 32 * 4);
    DNC(1, k_maxpool<KNC><<<dim3((P96 + 7) / 8), dim3(256), 0, stream>>>(x0, p, NB, 96, 96, 32, cptr));
    run_vgg(p, 32, 64, 64, 96, 96, getb(2), x1); }               // conv1_0
  resetT();
  { float* p = (float*)tall((size_t)P48 * 64 * 4);
    DNC(2, k_maxpool<KNC><<<dim3((P48 + 7) / 8), dim3(256), 0, stream>>>(x1, p, NB, 48, 48, 64, cptr));
    run_vgg(p, 64, 128, 128, 48, 48, getb(4), x2); }             // conv2_0
  resetT();
  { float* p = (float*)tall((size_t)P24 * 128 * 4);
    DNC(4, k_maxpool<KNC><<<dim3((P24 + 7) / 8), dim3(256), 0, stream>>>(x2, p, NB, 24, 24, 128, cptr));
    run_vgg(p, 128, 256, 256, 24, 24, getb(6), x3); }            // conv3_0
  resetT();
  { float* p = (float*)tall((size_t)P12 * 256 * 4);
    DNC(8, k_maxpool<KNC><<<dim3((P12 + 7) / 8), dim3(256), 0, stream>>>(x3, p, NB, 12, 12, 256, cptr));
    run_vgg(p, 256, 512, 512, 12, 12, getb(8), x4); }            // conv4_0

  // decoder
  resetT();
  { float* u = (float*)tall((size_t)P24 * 512 * 4);
    DNC(16, k_upsample<KNC><<<dim3((P24 + 7) / 8), dim3(256), 0, stream>>>(x4, u, NB, 12, 12, 512, cptr));
    float* cc = (float*)tall((size_t)P24 * 768 * 4);
    run_cat(x3, u, P24, 256, 512, cc);
    run_vgg(cc, 768, 256, 256, 24, 24, getb(7), x3n); }          // conv3_1
  resetT();
  { float* u = (float*)tall((size_t)P48 * 256 * 4);
    DNC(8, k_upsample<KNC><<<dim3((P48 + 7) / 8), dim3(256), 0, stream>>>(x3n, u, NB, 24, 24, 256, cptr));
    float* cc = (float*)tall((size_t)P48 * 384 * 4);
    run_cat(x2, u, P48, 128, 256, cc);
    run_vgg(cc, 384, 128, 128, 48, 48, getb(5), x2n); }          // conv2_2
  resetT();
  { float* u = (float*)tall((size_t)P96 * 128 * 4);
    DNC(4, k_upsample<KNC><<<dim3((P96 + 7) / 8), dim3(256), 0, stream>>>(x2n, u, NB, 48, 48, 128, cptr));
    float* cc = (float*)tall((size_t)P96 * 192 * 4);
    run_cat(x1, u, P96, 64, 128, cc);
    run_vgg(cc, 192, 64, 64, 96, 96, getb(3), x1n); }            // conv1_3
  resetT();
  { float* u = (float*)tall((size_t)P192 * 64 * 4);
    DNC(2, k_upsample<KNC><<<dim3((P192 + 7) / 8), dim3(256), 0, stream>>>(x1n, u, NB, 96, 96, 64, cptr));
    float* cc = (float*)tall((size_t)P192 * 96 * 4);
    run_cat(x0, u, P192, 32, 64, cc);
    run_vgg(cc, 96, 32, 32, 192, 192, getb(1), x0n); }           // conv0_4

  // final 1x1 hyperbolic conv (scale = beta(32)/beta(32) = 1) + logmap0 -> NCHW
  resetT();
  { float* tf = (float*)tall((size_t)P192 * 32 * 4);
    DNC(1, k_logmap0<KNC><<<dim3((P192 + 7) / 8), dim3(256), 0, stream>>>(x0n, tf, P192, 32, cptr));
    float* fb = (float*)tall((size_t)P192 * 4 * 4);
    run_fc(tf, final_z, final_r, 32, 4, 1, 0, 192, 192, 1.0f, fb);
    k_final_out<<<dim3((P192 + 255) / 256), dim3(256), 0, stream>>>(
        fb, (float*)d_out, NB, 192, 192, 4, cptr); }
}
